// ScaledDotProductAttention_47158740910630
// MI455X (gfx1250) — compile-verified, matrix operand order checked by reading, not run on hardware
//
#include <hip/hip_runtime.h>

// CDNA5 / gfx1250: wave32, bf16 WMMA, 320KB LDS per WGP.
// QT=32 query rows per block; each wave owns an EXCLUSIVE slice of K (stage 1)
// and V (stage 3) and reuses it for both 16-row query sub-tiles in registers,
// so K and V are each read from global exactly once per block. p_attn/out are
// streamed with non-temporal stores so the 536MB write-once stream does not
// evict the K/V reuse working set from the 192MB L2.

typedef __attribute__((ext_vector_type(16))) __bf16 v16bf;
typedef __attribute__((ext_vector_type(8)))  float  v8f;
typedef __attribute__((ext_vector_type(4)))  float  v4f;

#define S_LEN   2048
#define D_DIM   64
#define QT      32      // query rows per workgroup
#define THREADS 512     // 16 wave32s
#define NWAVES  16

// Build a 16(lane-dim) x 32(K) bf16 fragment where K is contiguous in memory.
// Caller folds the hi-lane (+8) K offset into p0.
// ISA layout: lo lanes: K = 0..7 (VGPR0..3), 16..23 (VGPR4..7); hi lanes +8.
__device__ __forceinline__ v16bf frag_k_contig(const float* p0) {
  v16bf f;
#pragma unroll
  for (int v = 0; v < 8; ++v) {
    const int k = (v < 4) ? (v * 2) : (16 + (v - 4) * 2);
    float2 d = *(const float2*)(p0 + k);
    f[2 * v + 0] = (__bf16)d.x;
    f[2 * v + 1] = (__bf16)d.y;
  }
  return f;
}

// Same fragment but K strides through memory (used for V: stride = D_DIM).
__device__ __forceinline__ v16bf frag_k_strided(const float* p0, int stride) {
  v16bf f;
#pragma unroll
  for (int v = 0; v < 8; ++v) {
    const int k = (v < 4) ? (v * 2) : (16 + (v - 4) * 2);
    f[2 * v + 0] = (__bf16)p0[(k + 0) * stride];
    f[2 * v + 1] = (__bf16)p0[(k + 1) * stride];
  }
  return f;
}

__global__ __launch_bounds__(THREADS)
void attn_softmax1_kernel(const float* __restrict__ q,
                          const float* __restrict__ k,
                          const float* __restrict__ v,
                          float* __restrict__ out,
                          float* __restrict__ p_out) {
  __shared__ float s_scores[QT * S_LEN];   // 256 KB score/prob strip
  __shared__ float s_out[QT * D_DIM];      // 8 KB PV accumulator
  __shared__ float s_part[THREADS];        // 2 KB reduction scratch
  __shared__ float s_rowmax[QT];
  __shared__ float s_rowinv[QT];

  const int t    = threadIdx.x;
  const int lane = t & 31;
  const int wave = t >> 5;                 // 0..15
  const int nlo  = lane & 15;
  const int hi   = lane >> 4;

  const int bh = blockIdx.x >> 6;          // 0..31  (B*H)
  const int qt = blockIdx.x & 63;          // 0..63  (S/32 query tiles)
  const int q0 = qt * QT;

  const float* qb  = q + (size_t)(bh * S_LEN + q0) * D_DIM;
  const float* kbp = k + (size_t)bh * S_LEN * D_DIM;
  const float* vbp = v + (size_t)bh * S_LEN * D_DIM;

  for (int i = t; i < QT * D_DIM; i += THREADS) s_out[i] = 0.0f;

  // ---------------- stage 1: S = scale * Q K^T ----------------
  // Each wave owns 8 key-tiles exclusively; Q A-fragments for BOTH 16-row
  // query sub-tiles stay in registers so each K row is loaded once.
  const v16bf aq00 = frag_k_contig(qb + (nlo)      * D_DIM + 0  + hi * 8);
  const v16bf aq01 = frag_k_contig(qb + (nlo)      * D_DIM + 32 + hi * 8);
  const v16bf aq10 = frag_k_contig(qb + (16 + nlo) * D_DIM + 0  + hi * 8);
  const v16bf aq11 = frag_k_contig(qb + (16 + nlo) * D_DIM + 32 + hi * 8);
  const float scale = 0.125f;              // 1/sqrt(64)

  const int kt_end = wave * 8 + 8;
  for (int kt = wave * 8; kt < kt_end; ++kt) {
    const int key0 = kt * 16;
    const float* krow = kbp + (size_t)(key0 + nlo) * D_DIM + hi * 8;
    if (kt + 1 < kt_end)
      __builtin_prefetch(kbp + (size_t)(key0 + 16 + nlo) * D_DIM, 0, 0);
    // B fragments (K^T): lane-dim = key index, K-dim = head dim; loaded ONCE,
    // consumed by both query sub-tiles.
    const v16bf b0 = frag_k_contig(krow + 0);
    const v16bf b1 = frag_k_contig(krow + 32);
    v8f c0 = {}, c1 = {};
    c0 = __builtin_amdgcn_wmma_f32_16x16x32_bf16(false, aq00, false, b0, (short)0, c0, false, false);
    c0 = __builtin_amdgcn_wmma_f32_16x16x32_bf16(false, aq01, false, b1, (short)0, c0, false, false);
    c1 = __builtin_amdgcn_wmma_f32_16x16x32_bf16(false, aq10, false, b0, (short)0, c1, false, false);
    c1 = __builtin_amdgcn_wmma_f32_16x16x32_bf16(false, aq11, false, b1, (short)0, c1, false, false);
#pragma unroll
    for (int r = 0; r < 8; ++r) {
      const int m = r + hi * 8;            // C layout: VGPR r -> row, lane -> col
      s_scores[(m)      * S_LEN + key0 + nlo] = c0[r] * scale;
      s_scores[(16 + m) * S_LEN + key0 + nlo] = c1[r] * scale;
    }
  }
  __syncthreads();

  // ---------------- stage 2: softmax-one + p_attn store ----------------
  const int row = t >> 4;                  // 0..31, 16 threads per row
  const int sub = t & 15;
  const int j0  = sub * (S_LEN / 16);      // 128 contiguous cols each
  float* srow = &s_scores[row * S_LEN];

  float lm = -3.402823466e38f;
  for (int j = 0; j < 128; j += 4) {
    v4f d = *(v4f*)&srow[j0 + j];
    lm = fmaxf(lm, fmaxf(fmaxf(d.x, d.y), fmaxf(d.z, d.w)));
  }
  s_part[t] = lm;
  __syncthreads();
  if (sub == 0) {
    float m = s_part[row * 16];
    for (int i = 1; i < 16; ++i) m = fmaxf(m, s_part[row * 16 + i]);
    s_rowmax[row] = m;
  }
  __syncthreads();
  const float rm = s_rowmax[row];

  // pass 2: e = exp(s - rm); write back in place, accumulate row sum
  float lsum = 0.0f;
  for (int j = 0; j < 128; j += 4) {
    v4f d = *(v4f*)&srow[j0 + j];
    v4f e;
    e.x = __expf(d.x - rm);
    e.y = __expf(d.y - rm);
    e.z = __expf(d.z - rm);
    e.w = __expf(d.w - rm);
    *(v4f*)&srow[j0 + j] = e;
    lsum += e.x + e.y + e.z + e.w;
  }
  s_part[t] = lsum;
  __syncthreads();
  if (sub == 0) {
    float s = 0.0f;
    for (int i = 0; i < 16; ++i) s += s_part[row * 16 + i];
    s_rowinv[row] = 1.0f / (1.0f + s);     // off-by-one softmax denominator
  }
  __syncthreads();
  const float inv = s_rowinv[row];

  // pass 3: p = e * inv; keep in LDS for PV, stream p_attn to HBM with
  // non-temporal stores (write-once data must not evict K/V from L2).
  float* prow = p_out + (size_t)(bh * S_LEN + q0 + row) * S_LEN;
  for (int j = 0; j < 128; j += 4) {
    v4f e = *(v4f*)&srow[j0 + j];
    v4f p;
    p.x = e.x * inv;
    p.y = e.y * inv;
    p.z = e.z * inv;
    p.w = e.w * inv;
    *(v4f*)&srow[j0 + j] = p;
    __builtin_nontemporal_store(p, (v4f*)&prow[j0 + j]);
  }
  __syncthreads();

  // ---------------- stage 3: out = P V ----------------
  // Each wave owns an exclusive K-range of 128 (4 chunks of 32); each V
  // B-fragment is loaded ONCE and used for both query sub-tiles.
#pragma unroll
  for (int nt = 0; nt < 4; ++nt) {         // 4 column tiles cover D=64
    v8f acc0 = {}, acc1 = {};
    for (int ci = 0; ci < 4; ++ci) {
      const int kb2 = wave * 128 + ci * 32;
      // B fragment from V: lane-dim = output col, K strides by D_DIM
      const float* vp = vbp + (size_t)(kb2 + hi * 8) * D_DIM + nt * 16 + nlo;
      const v16bf bv = frag_k_strided(vp, D_DIM);
      // A fragments from P in LDS for both sub-tiles (bf16-convert on the fly)
      v16bf pa0, pa1;
#pragma unroll
      for (int vv = 0; vv < 8; ++vv) {
        const int kk = ((vv < 4) ? (vv * 2) : (16 + (vv - 4) * 2)) + hi * 8;
        float2 d0 = *(float2*)&s_scores[(nlo)      * S_LEN + kb2 + kk];
        float2 d1 = *(float2*)&s_scores[(16 + nlo) * S_LEN + kb2 + kk];
        pa0[2 * vv + 0] = (__bf16)d0.x;
        pa0[2 * vv + 1] = (__bf16)d0.y;
        pa1[2 * vv + 0] = (__bf16)d1.x;
        pa1[2 * vv + 1] = (__bf16)d1.y;
      }
      acc0 = __builtin_amdgcn_wmma_f32_16x16x32_bf16(false, pa0, false, bv, (short)0, acc0, false, false);
      acc1 = __builtin_amdgcn_wmma_f32_16x16x32_bf16(false, pa1, false, bv, (short)0, acc1, false, false);
    }
#pragma unroll
    for (int r = 0; r < 8; ++r) {
      const int m = r + hi * 8;
      atomicAdd(&s_out[(m)      * D_DIM + nt * 16 + nlo], acc0[r]);  // ds_add_f32
      atomicAdd(&s_out[(16 + m) * D_DIM + nt * 16 + nlo], acc1[r]);
    }
  }
  __syncthreads();

  // write 32x64 output tile, float4 per thread (non-temporal: write-once)
  {
    const int idx = t * 4;
    const int r = idx >> 6;
    const int d = idx & 63;
    v4f o = *(v4f*)&s_out[idx];
    __builtin_nontemporal_store(o, (v4f*)&out[(size_t)(bh * S_LEN + q0 + r) * D_DIM + d]);
  }
}

extern "C" void kernel_launch(void* const* d_in, const int* in_sizes, int n_in,
                              void* d_out, int out_size, void* d_ws, size_t ws_size,
                              hipStream_t stream) {
  (void)in_sizes; (void)n_in; (void)d_ws; (void)ws_size; (void)out_size;
  const float* q = (const float*)d_in[0];
  const float* k = (const float*)d_in[1];
  const float* v = (const float*)d_in[2];
  float* out   = (float*)d_out;                          // [B,H,S,D]
  float* p_out = out + (size_t)2 * 16 * S_LEN * D_DIM;   // [B,H,S,S] follows out
  const int blocks = 2 * 16 * (S_LEN / QT);              // 2048
  attn_softmax1_kernel<<<dim3(blocks), dim3(THREADS), 0, stream>>>(q, k, v, out, p_out);
}
